// BFLinear_4698694222253
// MI455X (gfx1250) — matrix operand order repaired and checked by reading, not compiled, not run on hardware
//
#include <hip/hip_runtime.h>
#include <stdint.h>

// ---------- types ----------
typedef __attribute__((ext_vector_type(16))) __bf16  v16bf;
typedef __attribute__((ext_vector_type(8)))  float   v8f;
typedef __attribute__((ext_vector_type(4)))  unsigned int v4u;

union ABFrag { v4u u[2]; v16bf v; };   // 32 bytes, trivially constructible

// ---------- CDNA5 async global->LDS helpers (ASYNCcnt-tracked) ----------
__device__ __forceinline__ void async_ld128(unsigned lds_off, unsigned long long gaddr) {
    asm volatile("global_load_async_to_lds_b128 %0, %1, off"
                 :: "v"(lds_off), "v"(gaddr) : "memory");
}
__device__ __forceinline__ void wait_async0() {   // all async done
    asm volatile("s_wait_asynccnt 0" ::: "memory");
}
__device__ __forceinline__ void wait_async4() {   // oldest stage (4 loads) done
    asm volatile("s_wait_asynccnt 4" ::: "memory");
}

// =====================================================================
// Pass 1: block-floating-point quantize fp32 -> bf16 (exact for BFP-8)
// One lane per element; each wave32 == one BFP group of 32 elements.
// =====================================================================
__global__ void bfp_quant_kernel(const float* __restrict__ in,
                                 uint16_t* __restrict__ out,
                                 long long n)
{
    long long i = (long long)blockIdx.x * blockDim.x + threadIdx.x;
    if (i >= n) return;
    float x = in[i];
    float a = fabsf(x);
    #pragma unroll
    for (int off = 16; off > 0; off >>= 1)
        a = fmaxf(a, __shfl_xor(a, off, 32));

    float q = 0.0f;
    if (a > 0.0f) {
        int e = (int)((__float_as_uint(a) >> 23) & 0xFF) - 127;  // floor(log2) for normals
        if (e < -119) e = -119;                                  // keep 2^(e-7) normal
        float scale = __uint_as_float((unsigned)(e + 120) << 23); // 2^(e-7)
        float inv   = __uint_as_float((unsigned)(134 - e) << 23); // 2^(7-e)
        q = rintf(x * inv);                                       // RTE == jnp.round
        q = fminf(fmaxf(q, -128.0f), 127.0f) * scale;
    }
    out[i] = (uint16_t)(__float_as_uint(q) >> 16);  // exact: <= 8 significant bits
}

// =====================================================================
// Pass 2: C = Aq(bf16, MxK) * Bq(bf16, NxK)^T + bias  via WMMA bf16
//   128x128 block tile, BK=32, 8 waves (2 in M x 4 in N),
//   3-stage circular LDS buffer filled by global_load_async_to_lds_b128,
//   branch-free steady-state loop + 2 peeled tail stages.
// =====================================================================
#define BM 128
#define BN 128
#define BK 32
#define LDSP 80                               // LDS row pitch (64B data + 16B pad)
#define STAGE_BYTES (BM * LDSP + BN * LDSP)   // 20480 per stage
#define NSTAGE 3                              // 60 KB total

__global__ __launch_bounds__(256)
void bfp_gemm_kernel(const uint16_t* __restrict__ A,   // M x K (bf16 bits)
                     const uint16_t* __restrict__ B,   // N x K (bf16 bits)
                     const float*    __restrict__ bias,
                     float*          __restrict__ C,   // M x N
                     int M, int N, int K)
{
    __shared__ v4u ldsbuf[NSTAGE * STAGE_BYTES / 16];

    const int tid   = threadIdx.x;
    const int lane  = tid & 31;
    const int wave  = tid >> 5;
    const int waveM = wave & 1;    // 2 waves along M
    const int waveN = wave >> 1;   // 4 waves along N
    const int h     = lane >> 4;   // lane half (WMMA layout)
    const int l16   = lane & 15;

    const int mBlock = blockIdx.y * BM;
    const int nBlock = blockIdx.x * BN;

    // ---- per-thread tile-fill addressing: 2 A chunks + 2 B chunks of 16B ----
    const int r = tid >> 2;        // row 0..63 (second chunk is row+64)
    const int c = tid & 3;         // 16B chunk within the 64B row
    unsigned long long ga0 = (unsigned long long)(const char*)(A + (size_t)(mBlock + r) * K + c * 8);
    unsigned long long ga1 = ga0 + (unsigned long long)64 * K * 2;
    unsigned long long gb0 = (unsigned long long)(const char*)(B + (size_t)(nBlock + r) * K + c * 8);
    unsigned long long gb1 = gb0 + (unsigned long long)64 * K * 2;

    // LDS byte addresses (flat addr[31:0] is the LDS byte address per ISA)
    const unsigned ldsBase = (unsigned)(uintptr_t)(void*)ldsbuf;
    const unsigned la0 = ldsBase + r * LDSP + c * 16;
    const unsigned la1 = la0 + 64 * LDSP;
    const unsigned lb0 = ldsBase + BM * LDSP + r * LDSP + c * 16;
    const unsigned lb1 = lb0 + 64 * LDSP;

    v8f zero = {};
    v8f acc[4][2];
    #pragma unroll
    for (int wm = 0; wm < 4; ++wm)
        #pragma unroll
        for (int wn = 0; wn < 2; ++wn)
            acc[wm][wn] = zero;

    // one K-step of WMMA work from LDS stage at byte offset `off`
    auto computeStage = [&](unsigned off) {
        const char* ldsA = (const char*)ldsbuf + off;
        const char* ldsB = (const char*)ldsbuf + off + BM * LDSP;

        ABFrag afrag[4];
        #pragma unroll
        for (int wm = 0; wm < 4; ++wm) {
            const char* rp = ldsA + (waveM * 64 + wm * 16 + l16) * LDSP;
            afrag[wm].u[0] = *(const v4u*)(rp + 16 * h);        // K[8h .. 8h+7]
            afrag[wm].u[1] = *(const v4u*)(rp + 32 + 16 * h);   // K[16+8h .. +7]
        }
        ABFrag bfrag[2];
        #pragma unroll
        for (int wn = 0; wn < 2; ++wn) {
            const char* rp = ldsB + (waveN * 32 + wn * 16 + l16) * LDSP;
            bfrag[wn].u[0] = *(const v4u*)(rp + 16 * h);        // K[8h .. 8h+7]
            bfrag[wn].u[1] = *(const v4u*)(rp + 32 + 16 * h);   // K[16+8h .. +7]
        }
        #pragma unroll
        for (int wm = 0; wm < 4; ++wm)
            #pragma unroll
            for (int wn = 0; wn < 2; ++wn)
                acc[wm][wn] = __builtin_amdgcn_wmma_f32_16x16x32_bf16(
                    false, afrag[wm].v, false, bfrag[wn].v,
                    (short)0, acc[wm][wn], false, false);
    };

    const int nIter = K / BK;

    // ---- prologue: async-fill stages 0 and 1 ----
    unsigned o0 = 0, o1 = STAGE_BYTES, o2 = 2 * STAGE_BYTES;  // rotating slots
    async_ld128(la0 + o0, ga0); async_ld128(la1 + o0, ga1);
    async_ld128(lb0 + o0, gb0); async_ld128(lb1 + o0, gb1);
    ga0 += BK * 2; ga1 += BK * 2; gb0 += BK * 2; gb1 += BK * 2;
    if (nIter > 1) {
        async_ld128(la0 + o1, ga0); async_ld128(la1 + o1, ga1);
        async_ld128(lb0 + o1, gb0); async_ld128(lb1 + o1, gb1);
        ga0 += BK * 2; ga1 += BK * 2; gb0 += BK * 2; gb1 += BK * 2;
    }

    // ---- steady state: branch-free, always 2 stages in flight ----
    #pragma unroll 3
    for (int it = 0; it < nIter - 2; ++it) {
        wait_async4();                         // stage `it` done (in-order loads)
        __syncthreads();
        // prefetch stage it+2 into slot o2 (its old contents were consumed
        // at iteration it-1, and the barrier above ordered that)
        async_ld128(la0 + o2, ga0); async_ld128(la1 + o2, ga1);
        async_ld128(lb0 + o2, gb0); async_ld128(lb1 + o2, gb1);
        ga0 += BK * 2; ga1 += BK * 2; gb0 += BK * 2; gb1 += BK * 2;

        computeStage(o0);
        unsigned t = o0; o0 = o1; o1 = o2; o2 = t;   // rotate slots
    }

    // ---- peeled tail: stage nIter-2 (one stage still in flight) ----
    if (nIter >= 2) {
        wait_async4();
        __syncthreads();
        computeStage(o0);
        unsigned t = o0; o0 = o1; o1 = o2; o2 = t;
    }
    // ---- peeled tail: final stage nIter-1 ----
    wait_async0();
    __syncthreads();
    computeStage(o0);

    // ---- epilogue: add bias, store fp32 (C/D layout: n=l16, m=v+8h) ----
    const int nBase = nBlock + waveN * 32;
    float bv[2];
    bv[0] = bias[nBase + l16];
    bv[1] = bias[nBase + 16 + l16];

    #pragma unroll
    for (int wm = 0; wm < 4; ++wm) {
        #pragma unroll
        for (int v = 0; v < 8; ++v) {
            int m = mBlock + waveM * 64 + wm * 16 + v + 8 * h;
            #pragma unroll
            for (int wn = 0; wn < 2; ++wn) {
                int n = nBase + wn * 16 + l16;
                C[(size_t)m * N + n] = acc[wm][wn][v] + bv[wn];
            }
        }
    }
}

// =====================================================================
// Launcher
// =====================================================================
extern "C" void kernel_launch(void* const* d_in, const int* in_sizes, int n_in,
                              void* d_out, int out_size, void* d_ws, size_t ws_size,
                              hipStream_t stream)
{
    const float* x    = (const float*)d_in[0];
    const float* w    = (const float*)d_in[1];
    const float* bias = (const float*)d_in[2];
    float* out = (float*)d_out;

    const long long xsz = in_sizes[0];       // M * K
    const long long wsz = in_sizes[1];       // N * K
    const int N = in_sizes[2];
    const int K = (int)(wsz / N);
    const int M = (int)(xsz / K);

    uint16_t* xq = (uint16_t*)d_ws;          // M*K bf16
    uint16_t* wq = xq + xsz;                 // N*K bf16

    bfp_quant_kernel<<<(unsigned)((xsz + 255) / 256), 256, 0, stream>>>(x, xq, xsz);
    bfp_quant_kernel<<<(unsigned)((wsz + 255) / 256), 256, 0, stream>>>(w, wq, wsz);

    dim3 grid(N / BN, M / BM);
    bfp_gemm_kernel<<<grid, 256, 0, stream>>>(xq, wq, bias, out, M, N, K);
}
